// SuperLoss3_18983755448981
// MI455X (gfx1250) — compile-verified, matrix operand order checked by reading, not run on hardware
//
#include <hip/hip_runtime.h>
#include <hip/hip_bf16.h>

typedef __attribute__((ext_vector_type(8)))  float  v8f;
typedef __attribute__((ext_vector_type(16))) __bf16 v16bf;
typedef __attribute__((ext_vector_type(8)))  __bf16 v8bf;
typedef __attribute__((ext_vector_type(4)))  __bf16 v4bf;

#define NB 8192
#define NC 128
#define NTILES (NB / 16)      // 512 column tiles
#define SPLIT 8               // column-chunk split for occupancy
#define CHUNK (NTILES / SPLIT)
#define BPAD 136              // padded LDS row stride (bf16): optimal bank spread

#if __has_builtin(__builtin_amdgcn_global_load_async_to_lds_b128)
#define USE_ASYNC_LDS 1
typedef int v4i_pay __attribute__((vector_size(16)));               // 128-bit payload
typedef __attribute__((address_space(1))) v4i_pay gpay_t;           // global int4
typedef __attribute__((address_space(3))) v4i_pay lpay_t;           // LDS int4
__device__ __forceinline__ void async_cp16(const void* g, void* l) {
  __builtin_amdgcn_global_load_async_to_lds_b128((gpay_t*)(void*)g,
                                                 (lpay_t*)l, 0, 0);
}
__device__ __forceinline__ void wait_async0() {
#if __has_builtin(__builtin_amdgcn_s_wait_asynccnt)
  __builtin_amdgcn_s_wait_asynccnt(0);
#else
  asm volatile("s_wait_asynccnt 0x0" ::: "memory");
#endif
}
#else
#define USE_ASYNC_LDS 0
#endif

__device__ __forceinline__ float wred_sum(float v) {
#pragma unroll
  for (int off = 16; off >= 1; off >>= 1) v += __shfl_xor(v, off, 32);
  return v;
}
__device__ __forceinline__ float wred_max(float v) {
#pragma unroll
  for (int off = 16; off >= 1; off >>= 1) v = fmaxf(v, __shfl_xor(v, off, 32));
  return v;
}
// values kept non-negative -> IEEE bits order like ints
__device__ __forceinline__ void atomicMaxPosF(float* a, float v) {
  atomicMax((int*)a, __float_as_int(v));
}
__device__ __forceinline__ void atomicMinPosF(float* a, float v) {
  atomicMin((int*)a, __float_as_int(v));
}

// ---------------------------------------------------------------------------
// Split-bf16 decomposition: x = hi + lo (hi = rne(x), lo = rne(x - hi)).
__global__ __launch_bounds__(256) void convert_kernel(
    const float* __restrict__ x, __bf16* __restrict__ xhi, __bf16* __restrict__ xlo) {
  const int i = (blockIdx.x * 256 + threadIdx.x) * 4;
  const float4 v = *reinterpret_cast<const float4*>(x + i);
  const __bf16 h0 = (__bf16)v.x, h1 = (__bf16)v.y, h2 = (__bf16)v.z, h3 = (__bf16)v.w;
  const v4bf hv = {h0, h1, h2, h3};
  const v4bf lv = {(__bf16)(v.x - (float)h0), (__bf16)(v.y - (float)h1),
                   (__bf16)(v.z - (float)h2), (__bf16)(v.w - (float)h3)};
  *reinterpret_cast<v4bf*>(xhi + i) = hv;
  *reinterpret_cast<v4bf*>(xlo + i) = lv;
}

// ---------------------------------------------------------------------------
__global__ __launch_bounds__(256) void init_kernel(float* __restrict__ maxg,
                                                   float* __restrict__ ming,
                                                   float* __restrict__ acc) {
  const int i = blockIdx.x * 256 + threadIdx.x;
  maxg[i] = 0.0f;        // squared distances clamped >= 0
  ming[i] = 3.0e38f;
  if (i < 2) acc[i] = 0.0f;
}

// ---------------------------------------------------------------------------
// One wave32 per row: sq[row] = ||x_row||^2 (exact f32) ; CE contribution.
__global__ __launch_bounds__(256) void row_stats_kernel(
    const float* __restrict__ x, const int* __restrict__ tgt,
    float* __restrict__ sq, float* __restrict__ acc) {
  const int row  = blockIdx.x * 8 + (threadIdx.x >> 5);
  const int lane = threadIdx.x & 31;
  const float4 v = reinterpret_cast<const float4*>(x + (size_t)row * NC)[lane];

  float s = v.x * v.x + v.y * v.y + v.z * v.z + v.w * v.w;
  s = wred_sum(s);
  float mx = fmaxf(fmaxf(v.x, v.y), fmaxf(v.z, v.w));
  mx = wred_max(mx);
  float se = expf(v.x - mx) + expf(v.y - mx) + expf(v.z - mx) + expf(v.w - mx);
  se = wred_sum(se);

  if (lane == 0) {
    sq[row] = s;
    const float tl = x[(size_t)row * NC + tgt[row]];
    atomicAdd(acc + 0, mx + logf(se) - tl);
  }
}

// ---------------------------------------------------------------------------
// Fused Gram via split-bf16 WMMA (hi*hi + hi*lo + lo*hi), K=128 in 4 steps of 32.
// Hardest pos/neg tracked on SQUARED distances (sqrt deferred: monotone).
// B tiles (bf16 hi+lo, 4KB each) double-buffered in LDS via async-to-LDS DMA.
__global__ __launch_bounds__(256) void gram_triplet_kernel(
    const __bf16* __restrict__ xhi, const __bf16* __restrict__ xlo,
    const int* __restrict__ tgt, const float* __restrict__ sq,
    float* __restrict__ maxg, float* __restrict__ ming) {
  __shared__ __bf16 bufh[2][16 * BPAD];
  __shared__ __bf16 bufl[2][16 * BPAD];

  const int lane  = threadIdx.x & 31;
  const int wave  = threadIdx.x >> 5;
  const int agrp  = blockIdx.x / SPLIT;           // 0..63
  const int chnk  = blockIdx.x % SPLIT;           // 0..SPLIT-1
  const int i0    = (agrp * 8 + wave) * 16;       // anchor tile base row
  const int tjBeg = chnk * CHUNK;
  const int nsub  = lane & 15;                    // M (A) / N (B,C)
  const int g8    = (lane >> 4) * 8;              // K sub-base for 16-bit layout

  // cooperative staging map: thread t -> row r = t>>4, 8-elem chunk c16 = (t&15)*8
  const int sr = threadIdx.x >> 4, sc = (threadIdx.x & 15) * 8;

  // ---- build A fragments: 4 K-steps, hi+lo, per 16-bit A VGPR striping ----
  v16bf ahi[4], alo[4];
  {
    const __bf16* arh = xhi + (size_t)(i0 + nsub) * NC;
    const __bf16* arl = xlo + (size_t)(i0 + nsub) * NC;
#pragma unroll
    for (int s = 0; s < 4; ++s) {
      const v8bf h0 = *reinterpret_cast<const v8bf*>(arh + 32 * s + g8);
      const v8bf h1 = *reinterpret_cast<const v8bf*>(arh + 32 * s + 16 + g8);
      const v8bf l0 = *reinterpret_cast<const v8bf*>(arl + 32 * s + g8);
      const v8bf l1 = *reinterpret_cast<const v8bf*>(arl + 32 * s + 16 + g8);
#pragma unroll
      for (int j = 0; j < 8; ++j) {
        ahi[s][j] = h0[j]; ahi[s][8 + j] = h1[j];
        alo[s][j] = l0[j]; alo[s][8 + j] = l1[j];
      }
    }
  }

  const int mbase = i0 + 8 * (lane >> 4);
  float sqi[8]; int ti[8];
#pragma unroll
  for (int vv = 0; vv < 8; ++vv) { sqi[vv] = sq[mbase + vv]; ti[vv] = tgt[mbase + vv]; }

  float maxap[8], minan[8];
#pragma unroll
  for (int vv = 0; vv < 8; ++vv) { maxap[vv] = 0.0f; minan[vv] = 3.0e38f; }

  // ---- prologue: stage first tile (hi+lo) into buffer 0 ----
  {
    const int j0 = tjBeg * 16;
#if USE_ASYNC_LDS
    async_cp16(xhi + (size_t)(j0 + sr) * NC + sc, &bufh[0][sr * BPAD + sc]);
    async_cp16(xlo + (size_t)(j0 + sr) * NC + sc, &bufl[0][sr * BPAD + sc]);
    wait_async0();
#else
    const v8bf th = *reinterpret_cast<const v8bf*>(xhi + (size_t)(j0 + sr) * NC + sc);
    const v8bf tl = *reinterpret_cast<const v8bf*>(xlo + (size_t)(j0 + sr) * NC + sc);
    *reinterpret_cast<v8bf*>(&bufh[0][sr * BPAD + sc]) = th;
    *reinterpret_cast<v8bf*>(&bufl[0][sr * BPAD + sc]) = tl;
#endif
    __syncthreads();
  }

  for (int t = 0; t < CHUNK; ++t) {
    const int tj  = tjBeg + t;
    const int cur = t & 1;
    const int j0  = tj * 16;
    const bool havenext = (t + 1 < CHUNK);
    const int jn = (tj + 1) * 16;

#if USE_ASYNC_LDS
    if (havenext) {   // async DMA next tile while we crunch this one
      async_cp16(xhi + (size_t)(jn + sr) * NC + sc, &bufh[cur ^ 1][sr * BPAD + sc]);
      async_cp16(xlo + (size_t)(jn + sr) * NC + sc, &bufl[cur ^ 1][sr * BPAD + sc]);
    }
#else
    v8bf nh = {}, nl = {};
    if (havenext) {
      nh = *reinterpret_cast<const v8bf*>(xhi + (size_t)(jn + sr) * NC + sc);
      nl = *reinterpret_cast<const v8bf*>(xlo + (size_t)(jn + sr) * NC + sc);
    }
#endif

    // ---- B fragments from LDS (b128 reads, optimal bank spread) + WMMA ----
    const __bf16* bh = &bufh[cur][nsub * BPAD + g8];
    const __bf16* bl = &bufl[cur][nsub * BPAD + g8];
    v8f cHH = {}, cHL = {}, cLH = {};
#pragma unroll
    for (int s = 0; s < 4; ++s) {
      const v8bf bh0 = *reinterpret_cast<const v8bf*>(bh + 32 * s);
      const v8bf bh1 = *reinterpret_cast<const v8bf*>(bh + 32 * s + 16);
      const v8bf bl0 = *reinterpret_cast<const v8bf*>(bl + 32 * s);
      const v8bf bl1 = *reinterpret_cast<const v8bf*>(bl + 32 * s + 16);
      v16bf bhi, blo;
#pragma unroll
      for (int j = 0; j < 8; ++j) {
        bhi[j] = bh0[j]; bhi[8 + j] = bh1[j];
        blo[j] = bl0[j]; blo[8 + j] = bl1[j];
      }
      cHH = __builtin_amdgcn_wmma_f32_16x16x32_bf16(false, ahi[s], false, bhi,
                                                    (short)0, cHH, false, false);
      cHL = __builtin_amdgcn_wmma_f32_16x16x32_bf16(false, ahi[s], false, blo,
                                                    (short)0, cHL, false, false);
      cLH = __builtin_amdgcn_wmma_f32_16x16x32_bf16(false, alo[s], false, bhi,
                                                    (short)0, cLH, false, false);
    }

    const float sqn = sq[j0 + nsub];
    const int   tn  = tgt[j0 + nsub];
#pragma unroll
    for (int vv = 0; vv < 8; ++vv) {
      const float g  = cHH[vv] + (cHL[vv] + cLH[vv]);
      const float d2 = fmaxf(fmaf(-2.0f, g, sqi[vv] + sqn), 0.0f);  // squared dist
      const bool pos = (ti[vv] == tn);
      maxap[vv] = fmaxf(maxap[vv], pos ? d2 : 0.0f);                // branchless
      minan[vv] = fminf(minan[vv], pos ? 3.0e38f : d2);
    }

#if USE_ASYNC_LDS
    wait_async0();
    __syncthreads();
#else
    __syncthreads();                  // all waves done reading buf[cur^1]
    if (havenext) {
      *reinterpret_cast<v8bf*>(&bufh[cur ^ 1][sr * BPAD + sc]) = nh;
      *reinterpret_cast<v8bf*>(&bufl[cur ^ 1][sr * BPAD + sc]) = nl;
    }
    __syncthreads();                  // next tile visible to all waves
#endif
  }

  // reduce across the 16 N-lanes of each half-wave
#pragma unroll
  for (int vv = 0; vv < 8; ++vv) {
#pragma unroll
    for (int off = 8; off >= 1; off >>= 1) {
      maxap[vv] = fmaxf(maxap[vv], __shfl_xor(maxap[vv], off, 32));
      minan[vv] = fminf(minan[vv], __shfl_xor(minan[vv], off, 32));
    }
  }
  if (nsub == 0) {  // lanes 0 (M=0..7) and 16 (M=8..15)
#pragma unroll
    for (int vv = 0; vv < 8; ++vv) {
      atomicMaxPosF(&maxg[mbase + vv], maxap[vv]);
      atomicMinPosF(&ming[mbase + vv], minan[vv]);
    }
  }
}

// ---------------------------------------------------------------------------
// Deferred sqrt + 1e-12 floor (commutes with min/max), then margin-ReLU sum.
__global__ __launch_bounds__(256) void triplet_sum_kernel(
    const float* __restrict__ maxg, const float* __restrict__ ming,
    float* __restrict__ acc) {
  const int i = blockIdx.x * 256 + threadIdx.x;
  const float dap = sqrtf(fmaxf(maxg[i], 1e-12f));
  const float dan = sqrtf(fmaxf(ming[i], 1e-12f));
  float v = fmaxf(dap - dan + 0.3f, 0.0f);
  v = wred_sum(v);
  if ((threadIdx.x & 31) == 0) atomicAdd(acc + 1, v);
}

// ---------------------------------------------------------------------------
__global__ void finalize_kernel(const float* __restrict__ acc, float* __restrict__ out) {
  if (threadIdx.x == 0 && blockIdx.x == 0) {
    const float TAU = 4.852030263919617f;   // log(128)
    const float LAM = 0.25f;
    const float E   = 2.7182818284590452f;
    const float ce  = acc[0] / (float)NB;
    const float li  = acc[1] / (float)NB;

    const float z = 0.5f * fmaxf(-2.0f / E, (li - TAU) / LAM);
    const float p  = sqrtf(fmaxf(2.0f * (E * z + 1.0f), 0.0f));
    const float wn = -1.0f + p - p * p / 3.0f + (11.0f / 72.0f) * p * p * p;
    float w = (z < 0.0f) ? wn : log1pf(fmaxf(z, 0.0f));
#pragma unroll
    for (int it = 0; it < 10; ++it) {
      const float ew  = expf(w);
      const float f   = w * ew - z;
      const float wp1 = w + 1.0f;
      w = w - f / (ew * wp1 - (w + 2.0f) * f / (2.0f * wp1));
    }
    const float sigma = expf(-w);
    const float lg    = logf(sigma);
    out[0] = ((ce - TAU) * sigma + LAM * lg * lg) / (float)NB;
  }
}

// ---------------------------------------------------------------------------
extern "C" void kernel_launch(void* const* d_in, const int* in_sizes, int n_in,
                              void* d_out, int out_size, void* d_ws, size_t ws_size,
                              hipStream_t stream) {
  (void)in_sizes; (void)n_in; (void)out_size; (void)ws_size;
  const float* x   = (const float*)d_in[0];   // logits [8192,128] f32
  const int*   tgt = (const int*)d_in[1];     // targets [8192]

  // ws layout: xhi(2MB) | xlo(2MB) | sq | maxg | ming | acc   (~4.3MB total)
  __bf16* xhi = (__bf16*)d_ws;
  __bf16* xlo = xhi + (size_t)NB * NC;
  float*  sq   = (float*)(xlo + (size_t)NB * NC);
  float*  maxg = sq + NB;
  float*  ming = maxg + NB;
  float*  acc  = ming + NB;
  float*  out  = (float*)d_out;

  convert_kernel<<<(NB * NC) / 1024, 256, 0, stream>>>(x, xhi, xlo);
  init_kernel<<<NB / 256, 256, 0, stream>>>(maxg, ming, acc);
  row_stats_kernel<<<NB / 8, 256, 0, stream>>>(x, tgt, sq, acc);
  gram_triplet_kernel<<<64 * SPLIT, 256, 0, stream>>>(xhi, xlo, tgt, sq, maxg, ming);
  triplet_sum_kernel<<<NB / 256, 256, 0, stream>>>(maxg, ming, acc);
  finalize_kernel<<<1, 32, 0, stream>>>(acc, out);
}
